// GraphModule_82102594830998
// MI455X (gfx1250) — matrix operand order repaired and checked by reading, not compiled, not run on hardware
//
#include <hip/hip_runtime.h>

typedef __attribute__((ext_vector_type(16))) _Float16 v16h;
typedef __attribute__((ext_vector_type(8)))  float    v8f;

// Problem constants (match reference)
constexpr int  gN  = 20000;
constexpr int  gE  = 160000;
constexpr int  gET = gE + gN;     // edges + self loops = 180000
constexpr int  gB  = 512;
constexpr int  gH  = 4;
constexpr int  gC  = 256;
constexpr int  gD  = 1024;        // H*C
constexpr int  gIN = 77;
constexpr int  gK1 = 96;          // 77 padded to multiple of 32
constexpr float SLOPE_ATT = 0.2f;
constexpr float SLOPE_ACT = 0.01f;
constexpr float BN_EPS    = 1e-5f;
constexpr unsigned ORD_NEG_INF = 0x007FFFFFu; // f2ord(-inf)

// ---------- ordered-float encoding for atomic max ----------
__device__ __forceinline__ unsigned f2ord(float f) {
  unsigned u = __float_as_uint(f);
  return (u & 0x80000000u) ? ~u : (u | 0x80000000u);
}
__device__ __forceinline__ float ord2f(unsigned u) {
  return __uint_as_float((u & 0x80000000u) ? (u & 0x7FFFFFFFu) : ~u);
}
__device__ __forceinline__ float lrelu(float x, float s) { return x > 0.f ? x : s * x; }

// ---------- generic u32 fill ----------
__global__ void fill_u32(unsigned* __restrict__ p, unsigned v, long n) {
  long stride = (long)gridDim.x * blockDim.x;
  for (long i = (long)blockIdx.x * blockDim.x + threadIdx.x; i < n; i += stride) p[i] = v;
}

// ---------- pack A (row-major f32 [M,Ksrc]) into WMMA 16x32 f16 A-fragments ----------
// frag layout per ISA 7.12.2: lane L (m = tile*16 + L%16, half = L/16),
//   pair p<4 : kin = half*8 + 2p ; pair p>=4: kin = 16 + half*8 + 2(p-4)
__global__ void pack_a_f16(const float* __restrict__ src, _Float16* __restrict__ dst,
                           int M, int Ksrc, int Kpad) {
  int numKT = Kpad >> 5;
  long total = (long)M * Kpad;
  long stride = (long)gridDim.x * blockDim.x;
  for (long idx = (long)blockIdx.x * blockDim.x + threadIdx.x; idx < total; idx += stride) {
    int  j    = (int)(idx & 15);
    int  lane = (int)((idx >> 4) & 31);
    long tile = idx >> 9;
    int  kt   = (int)(tile % numKT);
    int  mt   = (int)(tile / numKT);
    int  half = lane >> 4;
    int  m    = (mt << 4) + (lane & 15);
    int  p    = j >> 1;
    int  kin  = (p < 4) ? (half * 8 + 2 * p) : (16 + half * 8 + 2 * (p - 4));
    int  k    = (kt << 5) + kin + (j & 1);
    float v = (k < Ksrc) ? src[(long)m * Ksrc + k] : 0.f;
    dst[idx] = (_Float16)v;
  }
}

// ---------- pack B (row-major f32 [Ksrc,Nn]) into WMMA 32x16 f16 B-fragments ----------
// lane L (col = tile*16 + L%16, half = L/16), pair p<4: kin = half*8 + 2p ; p>=4: kin = 16 + half*8 + 2(p-4)
__global__ void pack_b_f16(const float* __restrict__ src, _Float16* __restrict__ dst,
                           int Ksrc, int Kpad, int Nn) {
  int numNT = Nn >> 4;
  long total = (long)Kpad * Nn;
  long stride = (long)gridDim.x * blockDim.x;
  for (long idx = (long)blockIdx.x * blockDim.x + threadIdx.x; idx < total; idx += stride) {
    int  j    = (int)(idx & 15);
    int  lane = (int)((idx >> 4) & 31);
    long tile = idx >> 9;                 // kt*numNT + nt
    int  nt   = (int)(tile % numNT);
    int  kt   = (int)(tile / numNT);
    int  half = lane >> 4;
    int  col  = (nt << 4) + (lane & 15);
    int  kin  = (j < 8) ? (half * 8 + j) : (16 + half * 8 + (j - 8));   // table order, same as A
    int  k    = (kt << 5) + kin;
    float v = (k < Ksrc) ? src[(long)k * Nn + col] : 0.f;
    dst[idx] = (_Float16)v;
  }
}

// ---------- WMMA GEMM, 16x64 strip per wave (4 accumulators, A reused 4x) ----------
// C[M,Nn] f32 = Apk (frags) x Bpk (frags), optional column bias. Requires (Nn/16)%4==0.
__global__ void wmma_gemm_f16(const _Float16* __restrict__ Apk, const _Float16* __restrict__ Bpk,
                              float* __restrict__ Cout, const float* __restrict__ bias,
                              int M, int Nn, int numKT) {
  int numNT = Nn >> 4;
  int numNG = numNT >> 2;                     // groups of 4 consecutive n-tiles
  int numMT = M >> 4;
  int wave  = blockIdx.x * (blockDim.x >> 5) + (threadIdx.x >> 5);
  int lane  = threadIdx.x & 31;
  if (wave >= numMT * numNG) return;          // wave-uniform: EXEC stays all-ones
  int mt  = wave / numNG;
  int nt0 = (wave % numNG) << 2;

  const v16h* A = (const v16h*)Apk + ((long)mt * numKT) * 32 + lane;
  const v16h* B = (const v16h*)Bpk + (long)nt0 * 32 + lane;
  long bstep = (long)numNT * 32;

  v8f acc0 = {}, acc1 = {}, acc2 = {}, acc3 = {};
  for (int kt = 0; kt < numKT; ++kt) {
    v16h a = A[(long)kt * 32];
    const v16h* Bk = B + (long)kt * bstep;
    v16h b0 = Bk[0];
    v16h b1 = Bk[32];
    v16h b2 = Bk[64];
    v16h b3 = Bk[96];
    if (kt + 2 < numKT) {
      __builtin_prefetch((const void*)(A + (long)(kt + 2) * 32), 0, 1);
      __builtin_prefetch((const void*)(B + (long)(kt + 2) * bstep), 0, 1);
      __builtin_prefetch((const void*)(B + (long)(kt + 2) * bstep + 64), 0, 1);
    }
    acc0 = __builtin_amdgcn_wmma_f32_16x16x32_f16(false, a, false, b0, (short)0, acc0, false, false);
    acc1 = __builtin_amdgcn_wmma_f32_16x16x32_f16(false, a, false, b1, (short)0, acc1, false, false);
    acc2 = __builtin_amdgcn_wmma_f32_16x16x32_f16(false, a, false, b2, (short)0, acc2, false, false);
    acc3 = __builtin_amdgcn_wmma_f32_16x16x32_f16(false, a, false, b3, (short)0, acc3, false, false);
  }

  int colBase = (nt0 << 4) + (lane & 15);
  int row0    = (mt << 4) + ((lane >> 4) << 3);  // lanes 16..31 hold rows 8..15
  v8f accs[4] = {acc0, acc1, acc2, acc3};
#pragma unroll
  for (int q = 0; q < 4; ++q) {
    int col  = colBase + (q << 4);
    float bv = bias ? bias[col] : 0.f;
#pragma unroll
    for (int r = 0; r < 8; ++r)
      Cout[(long)(row0 + r) * Nn + col] = accs[q][r] + bv;
  }
}

// ---------- attention scores: one wave per (node, head) ----------
__global__ void att_scores(const float* __restrict__ hbuf, const float* __restrict__ a_src,
                           const float* __restrict__ a_dst, float* __restrict__ ssrc,
                           float* __restrict__ sdst) {
  int w    = blockIdx.x * (blockDim.x >> 5) + (threadIdx.x >> 5);
  int lane = threadIdx.x & 31;
  if (w >= gN * gH) return;
  int n = w >> 2, h = w & 3;
  const float* hp = hbuf + (long)n * gD + h * gC;
  const float* as = a_src + h * gC;
  const float* ad = a_dst + h * gC;
  float ss = 0.f, sd = 0.f;
  for (int j = lane; j < gC; j += 32) { float v = hp[j]; ss += v * as[j]; sd += v * ad[j]; }
  for (int o = 16; o; o >>= 1) { ss += __shfl_xor(ss, o, 32); sd += __shfl_xor(sd, o, 32); }
  if (lane == 0) { ssrc[w] = ss; sdst[w] = sd; }
}

// ---------- edge pass 1: segment max of leaky-relu'd logits ----------
__global__ void edge_max(const int* __restrict__ ei, const float* __restrict__ ssrc,
                         const float* __restrict__ sdst, unsigned* __restrict__ amax) {
  long stride = (long)gridDim.x * blockDim.x;
  for (long e = (long)blockIdx.x * blockDim.x + threadIdx.x; e < gET; e += stride) {
    int s = (e < gE) ? ei[e] : (int)(e - gE);
    int d = (e < gE) ? ei[gE + e] : (int)(e - gE);
#pragma unroll
    for (int h = 0; h < gH; ++h) {
      float a = lrelu(ssrc[s * gH + h] + sdst[d * gH + h], SLOPE_ATT);
      atomicMax(&amax[d * gH + h], f2ord(a));
    }
  }
}

// ---------- edge pass 2: exp + segment sum ----------
__global__ void edge_exp(const int* __restrict__ ei, const float* __restrict__ ssrc,
                         const float* __restrict__ sdst, const unsigned* __restrict__ amax,
                         float* __restrict__ denom, float* __restrict__ exb) {
  long stride = (long)gridDim.x * blockDim.x;
  for (long e = (long)blockIdx.x * blockDim.x + threadIdx.x; e < gET; e += stride) {
    int s = (e < gE) ? ei[e] : (int)(e - gE);
    int d = (e < gE) ? ei[gE + e] : (int)(e - gE);
#pragma unroll
    for (int h = 0; h < gH; ++h) {
      float a  = lrelu(ssrc[s * gH + h] + sdst[d * gH + h], SLOPE_ATT);
      float ex = __expf(a - ord2f(amax[d * gH + h]));
      exb[e * gH + h] = ex;
      atomicAdd(&denom[d * gH + h], ex);
    }
  }
}

// ---------- edge pass 3: aggregate messages (one 256-thread block per edge) ----------
__global__ void edge_aggr(const int* __restrict__ ei, const float* __restrict__ hbuf,
                          const float* __restrict__ exb, const float* __restrict__ denom,
                          float* __restrict__ aggr) {
  int t = threadIdx.x;           // 0..255, 4 feats each
  int head = t >> 6;             // 64 threads per head (256 feats)
  for (long e = blockIdx.x; e < gET; e += gridDim.x) {
    int s = (e < gE) ? ei[e] : (int)(e - gE);
    int d = (e < gE) ? ei[gE + e] : (int)(e - gE);
    float coef = exb[e * gH + head] / (denom[d * gH + head] + 1e-16f);
    const float4 hv = *(const float4*)(hbuf + (long)s * gD + t * 4);
    float* out = aggr + (long)d * gD + t * 4;
    atomicAdd(out + 0, hv.x * coef);
    atomicAdd(out + 1, hv.y * coef);
    atomicAdd(out + 2, hv.z * coef);
    atomicAdd(out + 3, hv.w * coef);
  }
}

// ---------- bias + BN(eval) + leaky relu, in place ----------
__global__ void bias_bn_lrelu(float* __restrict__ x, const float* __restrict__ b,
                              const float* __restrict__ g, const float* __restrict__ be,
                              const float* __restrict__ mean, const float* __restrict__ var) {
  long total = (long)gN * gD;
  long stride = (long)gridDim.x * blockDim.x;
  for (long i = (long)blockIdx.x * blockDim.x + threadIdx.x; i < total; i += stride) {
    int dcol = (int)(i & (gD - 1));
    float v = x[i] + b[dcol];
    v = (v - mean[dcol]) * rsqrtf(var[dcol] + BN_EPS) * g[dcol] + be[dcol];
    x[i] = lrelu(v, SLOPE_ACT);
  }
}

// ---------- global max pool by batch id ----------
__global__ void pool_max(const float* __restrict__ act, const int* __restrict__ batch,
                         unsigned* __restrict__ poolu) {
  long total = (long)gN * gD;
  long stride = (long)gridDim.x * blockDim.x;
  for (long i = (long)blockIdx.x * blockDim.x + threadIdx.x; i < total; i += stride) {
    int n = (int)(i >> 10);            // / gD
    int dcol = (int)(i & (gD - 1));
    atomicMax(&poolu[(long)batch[n] * gD + dcol], f2ord(act[i]));
  }
}

__global__ void pool_decode(const unsigned* __restrict__ poolu, float* __restrict__ pooled) {
  long total = (long)gB * gD;
  long stride = (long)gridDim.x * blockDim.x;
  for (long i = (long)blockIdx.x * blockDim.x + threadIdx.x; i < total; i += stride) {
    float v = ord2f(poolu[i]);
    pooled[i] = isfinite(v) ? v : 0.f;   // empty segments -> 0 (matches reference)
  }
}

// ------------------------------------------------------------------
extern "C" void kernel_launch(void* const* d_in, const int* in_sizes, int n_in,
                              void* d_out, int out_size, void* d_ws, size_t ws_size,
                              hipStream_t stream) {
  (void)in_sizes; (void)n_in; (void)out_size; (void)ws_size;
  const float* x      = (const float*)d_in[0];
  const int*   ei     = (const int*)d_in[1];
  const int*   batch  = (const int*)d_in[2];
  const float* W1     = (const float*)d_in[3];
  const float* a_src1 = (const float*)d_in[4];
  const float* a_dst1 = (const float*)d_in[5];
  const float* b1     = (const float*)d_in[6];
  const float* g1     = (const float*)d_in[7];
  const float* be1    = (const float*)d_in[8];
  const float* m1     = (const float*)d_in[9];
  const float* v1     = (const float*)d_in[10];
  const float* W2     = (const float*)d_in[11];
  const float* a_src2 = (const float*)d_in[12];
  const float* a_dst2 = (const float*)d_in[13];
  const float* b2     = (const float*)d_in[14];
  const float* g2     = (const float*)d_in[15];
  const float* be2    = (const float*)d_in[16];
  const float* m2     = (const float*)d_in[17];
  const float* v2     = (const float*)d_in[18];
  const float* fcW    = (const float*)d_in[19];
  const float* fcb    = (const float*)d_in[20];
  float* out = (float*)d_out;

  // ---- carve workspace ----
  char* w = (char*)d_ws;
  auto carve = [&](size_t bytes) { void* p = (void*)w; w += (bytes + 255) & ~(size_t)255; return p; };
  float*     hbuf   = (float*)    carve((size_t)gN * gD * 4);   // h1 then h2
  float*     aggr   = (float*)    carve((size_t)gN * gD * 4);   // aggregated + activated
  _Float16*  apk    = (_Float16*) carve((size_t)gN * gD * 2);   // packed A (max use)
  _Float16*  bpk    = (_Float16*) carve((size_t)gD * gD * 2);   // packed B (max use)
  float*     ssrc   = (float*)    carve((size_t)gN * gH * 4);
  float*     sdst   = (float*)    carve((size_t)gN * gH * 4);
  unsigned*  amax   = (unsigned*) carve((size_t)gN * gH * 4);
  float*     denom  = (float*)    carve((size_t)gN * gH * 4);
  float*     exb    = (float*)    carve((size_t)gET * gH * 4);
  unsigned*  poolu  = (unsigned*) carve((size_t)gB * gD * 4);
  float*     pooled = (float*)    carve((size_t)gB * gD * 4);

  auto blk = [](long n) { return (unsigned)((n + 255) / 256); };

  for (int layer = 0; layer < 2; ++layer) {
    const float* As = layer ? a_src2 : a_src1;
    const float* Ad = layer ? a_dst2 : a_dst1;
    const float* Bi = layer ? b2 : b1;
    const float* Gm = layer ? g2 : g1;
    const float* Be = layer ? be2 : be1;
    const float* Mn = layer ? m2 : m1;
    const float* Vr = layer ? v2 : v1;
    int Ksrc = layer ? gD : gIN;
    int Kpad = layer ? gD : gK1;
    const float* Ain = layer ? aggr : x;
    const float* Win = layer ? W2 : W1;

    // GEMM: h = Ain @ Win  (f16 WMMA, f32 accumulate, 16x64 strip per wave)
    pack_a_f16<<<blk((long)gN * Kpad), 256, 0, stream>>>(Ain, apk, gN, Ksrc, Kpad);
    pack_b_f16<<<blk((long)Kpad * gD), 256, 0, stream>>>(Win, bpk, Ksrc, Kpad, gD);
    {
      int waves = (gN / 16) * (gD / 64);
      wmma_gemm_f16<<<(waves + 7) / 8, 256, 0, stream>>>(apk, bpk, hbuf, nullptr,
                                                         gN, gD, Kpad / 32);
    }

    // attention scores
    att_scores<<<(gN * gH + 7) / 8, 256, 0, stream>>>(hbuf, As, Ad, ssrc, sdst);

    // edge softmax + aggregation
    fill_u32<<<blk(gN * gH), 256, 0, stream>>>(amax, ORD_NEG_INF, (long)gN * gH);
    fill_u32<<<blk(gN * gH), 256, 0, stream>>>((unsigned*)denom, 0u, (long)gN * gH);
    fill_u32<<<blk((long)gN * gD), 256, 0, stream>>>((unsigned*)aggr, 0u, (long)gN * gD);
    edge_max <<<blk(gET), 256, 0, stream>>>(ei, ssrc, sdst, amax);
    edge_exp <<<blk(gET), 256, 0, stream>>>(ei, ssrc, sdst, amax, denom, exb);
    edge_aggr<<<gET, 256, 0, stream>>>(ei, hbuf, exb, denom, aggr);

    // + bias, BN (eval), LeakyReLU(0.01)   (in place on aggr)
    bias_bn_lrelu<<<blk((long)gN * gD), 256, 0, stream>>>(aggr, Bi, Gm, Be, Mn, Vr);
  }

  // global max pool over batch -> [B, D]
  fill_u32<<<blk((long)gB * gD), 256, 0, stream>>>(poolu, ORD_NEG_INF, (long)gB * gD);
  pool_max  <<<blk((long)gN * gD), 256, 0, stream>>>(aggr, batch, poolu);
  pool_decode<<<blk((long)gB * gD), 256, 0, stream>>>(poolu, pooled);

  // fc: out = pooled @ fcW + fcb   (WMMA, bias fused in epilogue)
  pack_a_f16<<<blk((long)gB * gD), 256, 0, stream>>>(pooled, apk, gB, gD, gD);
  pack_b_f16<<<blk((long)gD * 256), 256, 0, stream>>>(fcW, bpk, gD, gD, 256);
  {
    int waves = (gB / 16) * (256 / 64);
    wmma_gemm_f16<<<(waves + 7) / 8, 256, 0, stream>>>(apk, bpk, out, fcb,
                                                       gB, 256, gD / 32);
  }
}
